// MoE_16260746182882
// MI455X (gfx1250) — matrix operand order repaired and checked by reading, not compiled, not run on hardware
//
#include <hip/hip_runtime.h>

// ---------------------------------------------------------------------------
// MoE with "plain" gating == single dense FFN through expert 1:
//   h   = relu(x @ W1[1] + b1[1])   [4096,2048]
//   out =       h @ W2[1] + b2[1]   [4096,1024]  (gate == 1.0f exactly in f32)
// bf16 WMMA (v_wmma_f32_16x16x32_bf16), f32 accumulate,
// async global->LDS staging (ASYNCcnt) with double buffering, BK=64,
// branch-free steady-state loop (last iteration peeled).
// ---------------------------------------------------------------------------

typedef __bf16 bf16;
typedef __attribute__((ext_vector_type(16))) __bf16 v16bf;
typedef __attribute__((ext_vector_type(4)))  __bf16 v4bf;
typedef __attribute__((ext_vector_type(8)))  float  v8f;

// ---------------- f32 -> bf16 bulk convert (vectorized x4) -----------------
__global__ __launch_bounds__(256)
void cvt_f32_bf16(const float* __restrict__ src, bf16* __restrict__ dst, int n) {
    int i = (blockIdx.x * 256 + threadIdx.x) * 4;
    if (i < n) {
        float4 f = *(const float4*)(src + i);
        v4bf o = { (bf16)f.x, (bf16)f.y, (bf16)f.z, (bf16)f.w };
        *(v4bf*)(dst + i) = o;
    }
}

// ------------- tiled transpose + convert: src[R][C] -> dst[C][R] -----------
__global__ __launch_bounds__(256)
void transpose_cvt(const float* __restrict__ src, bf16* __restrict__ dst,
                   int R, int C) {
    __shared__ float t[32][33];
    const int bx = blockIdx.x * 32;   // column base in src
    const int by = blockIdx.y * 32;   // row base in src
    const int tx = threadIdx.x, ty = threadIdx.y;
#pragma unroll
    for (int i = 0; i < 32; i += 8)
        t[ty + i][tx] = src[(size_t)(by + ty + i) * C + (bx + tx)];
    __syncthreads();
#pragma unroll
    for (int i = 0; i < 32; i += 8)
        dst[(size_t)(bx + ty + i) * R + (by + tx)] = (bf16)t[tx][ty + i];
}

// ------------------- CDNA5 async global -> LDS (b128) ----------------------
__device__ __forceinline__ void async_ld_b128(unsigned lds_addr, const void* gaddr) {
    // VDST = per-lane LDS byte address, VADDR = 64-bit global address
    asm volatile("global_load_async_to_lds_b128 %0, %1, off"
                 :: "v"(lds_addr), "v"(gaddr)
                 : "memory");
}
template <int N>
__device__ __forceinline__ void wait_async() {
    asm volatile("s_wait_asynccnt %0" :: "n"(N) : "memory");
}

// ------------------------- bf16 WMMA GEMM ----------------------------------
// C[M][N] = op(A[M][K] @ Bt[N][K]^T + bias[N]); op = relu if RELU.
// Block: 128 threads = 4 waves (2x2), tile BM=128 x BN=128, BK=64.
// Each wave: 4x4 tiles of 16x16 (64 rows x 64 cols), 128 acc VGPRs.
// Double-buffered LDS filled with global_load_async_to_lds_b128 (16/tile/thread).
template <bool RELU, typename OutT>
__global__ __launch_bounds__(128)
void gemm_bf16_wmma(const bf16* __restrict__ A, const bf16* __restrict__ Bt,
                    const float* __restrict__ bias, OutT* __restrict__ Cc,
                    int M, int N, int K) {
    constexpr int BM = 128, BN = 128, BK = 64, LDT = 72;  // 144B rows: 16B aligned, conflict-free
    constexpr int TILEB = BM * LDT * 2;                   // bytes per buffer (A == B here)
    __shared__ __align__(16) bf16 As[2][BM * LDT];
    __shared__ __align__(16) bf16 Bs[2][BN * LDT];

    const int tid    = threadIdx.x;
    const int wave   = tid >> 5;
    const int lane   = tid & 31;
    const int wm     = wave >> 1;        // 0..1
    const int wn     = wave & 1;         // 0..1
    const int laneLo = lane & 15;
    const int laneHi = lane >> 4;        // 0..1

    const int m0 = blockIdx.y * BM;
    const int n0 = blockIdx.x * BN;

    // LDS byte offsets (low 32 bits of the generic shared-pointer = LDS offset)
    const unsigned asBase = (unsigned)(uintptr_t)&As[0][0];
    const unsigned bsBase = (unsigned)(uintptr_t)&Bs[0][0];

    // global->LDS: 1024 chunks of 8 bf16 per 128x64 tile; 8 chunks/thread/operand
    // chunk c: row = c>>3, col = (c&7)*8
    const auto issue = [&](int k0, int buf) {
        const unsigned ab = asBase + (unsigned)buf * TILEB;
        const unsigned bb = bsBase + (unsigned)buf * TILEB;
#pragma unroll
        for (int j = 0; j < 8; ++j) {
            const int c  = tid + j * 128;
            const int r  = c >> 3;
            const int cc = (c & 7) * 8;
            async_ld_b128(ab + (unsigned)(r * LDT + cc) * 2,
                          &A[(size_t)(m0 + r) * K + k0 + cc]);
            async_ld_b128(bb + (unsigned)(r * LDT + cc) * 2,
                          &Bt[(size_t)(n0 + r) * K + k0 + cc]);
        }
    };

    v8f acc[4][4] = {};
    union Frag { v16bf v; uint4 q[2]; };

    // 2 x 16 WMMAs per K-tile (kk = K-half within the 64-wide tile)
    const auto compute = [&](int buf) {
#pragma unroll
        for (int kk = 0; kk < 2; ++kk) {
            const int kcol = kk * 32;
            // A fragment (16-bit A 16x32): lane<16 -> K 0..7 / 16..23,
            // lane>=16 -> K 8..15 / 24..31, row = lane&15.
            Frag a[4], b[4];
#pragma unroll
            for (int mi = 0; mi < 4; ++mi) {
                const int r  = wm * 64 + mi * 16 + laneLo;
                const int kb = kcol + laneHi * 8;
                a[mi].q[0] = *(const uint4*)&As[buf][r * LDT + kb];
                a[mi].q[1] = *(const uint4*)&As[buf][r * LDT + kb + 16];
            }
            // B fragment (32x16): lane holds N=lane&15, K = laneHi*8 .. +7 / +16 .. +23;
            // Bt tile is N-major so each 8 bf16 run is contiguous.
#pragma unroll
            for (int ni = 0; ni < 4; ++ni) {
                const int nn = wn * 64 + ni * 16 + laneLo;
                const int kb = kcol + laneHi * 8;
                b[ni].q[0] = *(const uint4*)&Bs[buf][nn * LDT + kb];
                b[ni].q[1] = *(const uint4*)&Bs[buf][nn * LDT + kb + 16];
            }
#pragma unroll
            for (int mi = 0; mi < 4; ++mi)
#pragma unroll
                for (int ni = 0; ni < 4; ++ni)
                    acc[mi][ni] = __builtin_amdgcn_wmma_f32_16x16x32_bf16(
                        false, a[mi].v, false, b[ni].v,
                        (short)0, acc[mi][ni], false, false);
        }
    };

    // ---- pipelined main loop, last iteration peeled (branch-free body) ----
    issue(0, 0);
    int buf = 0;
    for (int k0 = 0; k0 < K - BK; k0 += BK) {
        issue(k0 + BK, buf ^ 1);    // overlap next tile into other buffer
        wait_async<16>();           // 16 newest may remain in flight; tile k0 done
        __syncthreads();            // all waves' async data visible
        compute(buf);
        __syncthreads();            // done reading buf before it is refilled
        buf ^= 1;
    }
    wait_async<0>();
    __syncthreads();
    compute(buf);

    // C/D layout: VGPR v, lane -> row = v + (lane>>4)*8, col = lane&15
#pragma unroll
    for (int mi = 0; mi < 4; ++mi) {
        const int mbase = m0 + wm * 64 + mi * 16 + laneHi * 8;
#pragma unroll
        for (int ni = 0; ni < 4; ++ni) {
            const int n  = n0 + wn * 64 + ni * 16 + laneLo;
            const float bv = bias[n];
#pragma unroll
            for (int v = 0; v < 8; ++v) {
                float val = acc[mi][ni][v] + bv;
                if (RELU) val = val > 0.0f ? val : 0.0f;
                Cc[(size_t)(mbase + v) * N + n] = (OutT)val;
            }
        }
    }
}

// ---------------------------------------------------------------------------
extern "C" void kernel_launch(void* const* d_in, const int* in_sizes, int n_in,
                              void* d_out, int out_size, void* d_ws, size_t ws_size,
                              hipStream_t stream) {
    (void)in_sizes; (void)n_in; (void)out_size; (void)ws_size;

    constexpr int Bq = 4, S = 1024, D = 1024, O = 1024, H = 2048;
    constexpr int M = Bq * S;  // 4096 tokens

    const float* x  = (const float*)d_in[0];
    const float* W1 = (const float*)d_in[3];
    const float* b1 = (const float*)d_in[4];
    const float* W2 = (const float*)d_in[5];
    const float* b2 = (const float*)d_in[6];

    // expert 1 slices
    const float* W1e = W1 + (size_t)1 * D * H;
    const float* b1e = b1 + (size_t)1 * H;
    const float* W2e = W2 + (size_t)1 * H * O;
    const float* b2e = b2 + (size_t)1 * O;

    // workspace: xb[M*D] | W1t[H*D] | W2t[O*H] | h[M*H]  (bf16) = 32 MB
    char* ws = (char*)d_ws;
    bf16* xb   = (bf16*)ws;
    bf16* W1t  = (bf16*)(ws + (size_t)M * D * 2);
    bf16* W2t  = (bf16*)(ws + (size_t)(M * D + H * D) * 2);
    bf16* hbuf = (bf16*)(ws + (size_t)(M * D + H * D + O * H) * 2);

    // 1) x -> bf16
    cvt_f32_bf16<<<(M * D / 4 + 255) / 256, 256, 0, stream>>>(x, xb, M * D);
    // 2) W1[1]: [D][H] -> bf16 [H][D]
    transpose_cvt<<<dim3(H / 32, D / 32), dim3(32, 8), 0, stream>>>(W1e, W1t, D, H);
    // 3) W2[1]: [H][O] -> bf16 [O][H]
    transpose_cvt<<<dim3(O / 32, H / 32), dim3(32, 8), 0, stream>>>(W2e, W2t, H, O);
    // 4) h = relu(x @ W1 + b1)  -> bf16 [M][H]
    gemm_bf16_wmma<true, bf16>
        <<<dim3(H / 128, M / 128), 128, 0, stream>>>(xb, W1t, b1e, hbuf, M, H, D);
    // 5) out = h @ W2 + b2      -> f32 [M][O]
    gemm_bf16_wmma<false, float>
        <<<dim3(O / 128, M / 128), 128, 0, stream>>>(hbuf, W2t, b2e, (float*)d_out, M, O, H);
}
